// CommandModel_90993177133809
// MI455X (gfx1250) — matrix operand order repaired and checked by reading, not compiled, not run on hardware
//
#include <hip/hip_runtime.h>
#include <hip/hip_bf16.h>
#include <cstdint>

// ---------------------------------------------------------------------------
// Problem dimensions (fixed by the reference)
// ---------------------------------------------------------------------------
#define B_      64
#define H_      512
#define E_      256
#define V_      8192
#define G4H     2048      // 4*H
#define TSTEPS  200

typedef __attribute__((ext_vector_type(16))) __bf16 v16bf;
typedef __attribute__((ext_vector_type(8)))  float  v8f;

union FragAB {
    v16bf v;
    uint4 q[2];
};

// ---------------------------------------------------------------------------
// WMMA fragment loaders (bf16, wave32; contiguous k order, same for A and B)
// A: 16x32 (MxK). Lanes 0-15: row M=lane, K = [0..15].
//    Lanes 16-31: row M=lane-16, K = [16..31].
// B: 32x16 (KxN). Lanes 0-15: col N=lane, K = kk..kk+15 (16 contiguous bf16).
//    Lanes 16-31: col N=lane-16, K = kk+16..kk+31.
// ---------------------------------------------------------------------------
__device__ __forceinline__ void load_frag_A(const __bf16* __restrict__ base,
                                            int ldk, int m_base, int kk,
                                            int lane, FragAB& fa) {
    int row = m_base + (lane & 15);
    int kh  = (lane >> 4) * 16;
    const __bf16* p0 = base + (size_t)row * ldk + kk + kh;        // 8 bf16 = 16B
    const __bf16* p1 = base + (size_t)row * ldk + kk + 8 + kh;    // 8 bf16 = 16B
    fa.q[0] = *(const uint4*)p0;
    fa.q[1] = *(const uint4*)p1;
}

__device__ __forceinline__ void load_frag_B(const __bf16* __restrict__ base,
                                            int ldk, int n_base, int kk,
                                            int lane, FragAB& fb) {
    int col = n_base + (lane & 15);
    int k0  = kk + (lane >> 4) * 16;
    const __bf16* p = base + (size_t)col * ldk + k0;              // 16 bf16 = 32B
    fb.q[0] = ((const uint4*)p)[0];
    fb.q[1] = ((const uint4*)p)[1];
}

__device__ __forceinline__ v8f wmma_bf16(const FragAB& a, const FragAB& b, v8f c) {
    // 8 args: (neg_a, A, neg_b, B, c_mod, C, reuse_a, reuse_b)
    return __builtin_amdgcn_wmma_f32_16x16x32_bf16(false, a.v, false, b.v,
                                                   (short)0, c, false, false);
}

// ---------------------------------------------------------------------------
// One-time setup kernels
// ---------------------------------------------------------------------------
__global__ void f32_to_bf16_kernel(const float* __restrict__ in,
                                   __bf16* __restrict__ out, int n) {
    int i = blockIdx.x * blockDim.x + threadIdx.x;
    if (i < n) out[i] = (__bf16)in[i];
}

__global__ void init_state_kernel(const float* __restrict__ obs,
                                  float* __restrict__ c,
                                  __bf16* __restrict__ hbf) {
    int i = blockIdx.x * blockDim.x + threadIdx.x;   // B_*H_
    float v = obs[i];
    c[i]   = v;
    hbf[i] = (__bf16)v;
}

__global__ void init_x_kernel(const __bf16* __restrict__ embB,
                              __bf16* __restrict__ x) {
    int i = blockIdx.x * blockDim.x + threadIdx.x;   // B_*E_
    x[i] = embB[i & (E_ - 1)];                        // broadcast row 0 (START token)
}

// ---------------------------------------------------------------------------
// gates = x @ Wih^T + h @ Whh^T + b_ih + b_hh       [B_, 4H]
// grid = G4H/64 blocks of 256 threads; 8 waves = 4(M) x 2(N-group),
// each wave register-blocks 2 N-tiles (16x32 output, shared A fragment).
// ---------------------------------------------------------------------------
__global__ __launch_bounds__(256) void gates_kernel(
        const __bf16* __restrict__ x,      // [B_, E_]
        const __bf16* __restrict__ h,      // [B_, H_]
        const __bf16* __restrict__ Wih,    // [G4H, E_]
        const __bf16* __restrict__ Whh,    // [G4H, H_]
        const float*  __restrict__ b_ih,
        const float*  __restrict__ b_hh,
        float* __restrict__ gates)         // [B_, G4H]
{
    const int lane   = threadIdx.x & 31;
    const int wave   = threadIdx.x >> 5;
    const int m_base = (wave >> 1) * 16;                     // 0,16,32,48
    const int n_base = blockIdx.x * 64 + (wave & 1) * 32;    // wave covers [n_base, n_base+32)

    v8f acc0 = {}, acc1 = {};
    FragAB a, b0, b1;

    // x-part: K = E_
    for (int kk = 0; kk < E_; kk += 32) {
        __builtin_prefetch(Wih + (size_t)(n_base + (lane & 15)) * E_ + kk + 64, 0, 3);
        load_frag_A(x,   E_, m_base, kk, lane, a);
        load_frag_B(Wih, E_, n_base,      kk, lane, b0);
        load_frag_B(Wih, E_, n_base + 16, kk, lane, b1);
        acc0 = wmma_bf16(a, b0, acc0);
        acc1 = wmma_bf16(a, b1, acc1);
    }
    // h-part: K = H_
    for (int kk = 0; kk < H_; kk += 32) {
        __builtin_prefetch(Whh + (size_t)(n_base + (lane & 15)) * H_ + kk + 64, 0, 3);
        load_frag_A(h,   H_, m_base, kk, lane, a);
        load_frag_B(Whh, H_, n_base,      kk, lane, b0);
        load_frag_B(Whh, H_, n_base + 16, kk, lane, b1);
        acc0 = wmma_bf16(a, b0, acc0);
        acc1 = wmma_bf16(a, b1, acc1);
    }

    // D layout: lanes 0-15 -> N=lane, M=m_base+r; lanes 16-31 -> N=lane-16, M=m_base+8+r
    const int col0 = n_base + (lane & 15);
    const int col1 = col0 + 16;
    const int mrow = m_base + (lane >> 4) * 8;
    const float bias0 = b_ih[col0] + b_hh[col0];
    const float bias1 = b_ih[col1] + b_hh[col1];
#pragma unroll
    for (int r = 0; r < 8; ++r) {
        gates[(size_t)(mrow + r) * G4H + col0] = acc0[r] + bias0;
        gates[(size_t)(mrow + r) * G4H + col1] = acc1[r] + bias1;
    }
}

// ---------------------------------------------------------------------------
// LSTM pointwise update: c' = sig(f)*c + sig(i)*tanh(g); h = sig(o)*tanh(c')
// ---------------------------------------------------------------------------
__global__ __launch_bounds__(256) void lstm_update_kernel(
        const float* __restrict__ gates,   // [B_, G4H]
        float* __restrict__ c,             // [B_, H_]
        __bf16* __restrict__ hbf)          // [B_, H_]
{
    int idx = blockIdx.x * blockDim.x + threadIdx.x;   // B_*H_
    int b = idx >> 9;
    int u = idx & (H_ - 1);
    const float* g = gates + (size_t)b * G4H;
    float gi = g[u];
    float gf = g[H_ + u];
    float gg = g[2 * H_ + u];
    float go = g[3 * H_ + u];
    float si = 1.0f / (1.0f + __expf(-gi));
    float sf = 1.0f / (1.0f + __expf(-gf));
    float so = 1.0f / (1.0f + __expf(-go));
    float cn = sf * c[idx] + si * tanhf(gg);
    c[idx]   = cn;
    hbf[idx] = (__bf16)(so * tanhf(cn));
}

// ---------------------------------------------------------------------------
// logits = h @ Wout^T + b_out    [B_, V_], written straight into d_out slice
// grid = V_/64 blocks of 256 threads; same 2-N-tile register blocking
// ---------------------------------------------------------------------------
__global__ __launch_bounds__(256) void logits_kernel(
        const __bf16* __restrict__ h,      // [B_, H_]
        const __bf16* __restrict__ Wout,   // [V_, H_]
        const float*  __restrict__ b_out,
        float* __restrict__ logits)        // [B_, V_]
{
    const int lane   = threadIdx.x & 31;
    const int wave   = threadIdx.x >> 5;
    const int m_base = (wave >> 1) * 16;
    const int n_base = blockIdx.x * 64 + (wave & 1) * 32;

    v8f acc0 = {}, acc1 = {};
    FragAB a, b0, b1;
    for (int kk = 0; kk < H_; kk += 32) {
        __builtin_prefetch(Wout + (size_t)(n_base + (lane & 15)) * H_ + kk + 64, 0, 3);
        load_frag_A(h,    H_, m_base, kk, lane, a);
        load_frag_B(Wout, H_, n_base,      kk, lane, b0);
        load_frag_B(Wout, H_, n_base + 16, kk, lane, b1);
        acc0 = wmma_bf16(a, b0, acc0);
        acc1 = wmma_bf16(a, b1, acc1);
    }
    const int col0 = n_base + (lane & 15);
    const int col1 = col0 + 16;
    const int mrow = m_base + (lane >> 4) * 8;
    const float bias0 = b_out[col0];
    const float bias1 = b_out[col1];
#pragma unroll
    for (int r = 0; r < 8; ++r) {
        logits[(size_t)(mrow + r) * V_ + col0] = acc0[r] + bias0;
        logits[(size_t)(mrow + r) * V_ + col1] = acc1[r] + bias1;
    }
}

// ---------------------------------------------------------------------------
// Gumbel-argmax sampling + embedding gather. One block per batch row.
// (Counter-based hash RNG; JAX threefry is not bit-reproducible here.)
// ---------------------------------------------------------------------------
__device__ __forceinline__ uint32_t hash_u32(uint32_t x) {
    x ^= x >> 17; x *= 0xed5ad4bbu;
    x ^= x >> 11; x *= 0xac4c1b51u;
    x ^= x >> 15; x *= 0x31848babu;
    x ^= x >> 14;
    return x;
}

__global__ __launch_bounds__(256) void sample_gather_kernel(
        const float* __restrict__ logits,  // [B_, V_]
        const __bf16* __restrict__ embB,   // [V_, E_]
        int* __restrict__ tok_out,         // [B_] slice of tokens
        __bf16* __restrict__ x,            // [B_, E_] next input
        unsigned t)
{
    __shared__ float sval[256];
    __shared__ int   sidx[256];
    __shared__ int   stok;

    const int b   = blockIdx.x;
    const int tid = threadIdx.x;

    float best = -3.402823466e38f;
    int   bi   = 0;
    const uint32_t seed = (t * B_ + b) * (uint32_t)V_;
    for (int v = tid; v < V_; v += 256) {
        float l = logits[(size_t)b * V_ + v];
        uint32_t hsh = hash_u32(seed + (uint32_t)v);
        float u01  = ((float)hsh + 0.5f) * 2.3283064365386963e-10f; // (0,1)
        float gum  = -__logf(-__logf(u01));
        float s = l + gum;
        if (s > best) { best = s; bi = v; }
    }
    sval[tid] = best; sidx[tid] = bi;
    __syncthreads();
#pragma unroll
    for (int s = 128; s > 0; s >>= 1) {
        if (tid < s && sval[tid + s] > sval[tid]) {
            sval[tid] = sval[tid + s];
            sidx[tid] = sidx[tid + s];
        }
        __syncthreads();
    }
    if (tid == 0) { stok = sidx[0]; tok_out[b] = sidx[0]; }
    __syncthreads();

    // gather next x (E_ == blockDim.x == 256)
    const int tok = stok;
    x[(size_t)b * E_ + tid] = embB[(size_t)tok * E_ + tid];
}

// ---------------------------------------------------------------------------
// Host-side orchestration (graph-capture safe: launches only, on `stream`)
// ---------------------------------------------------------------------------
extern "C" void kernel_launch(void* const* d_in, const int* in_sizes, int n_in,
                              void* d_out, int out_size, void* d_ws, size_t ws_size,
                              hipStream_t stream) {
    const float* obs  = (const float*)d_in[0];   // [B_, H_]
    const float* emb  = (const float*)d_in[1];   // [V_, E_]
    const float* Wih  = (const float*)d_in[2];   // [G4H, E_]
    const float* Whh  = (const float*)d_in[3];   // [G4H, H_]
    const float* bih  = (const float*)d_in[4];   // [G4H]
    const float* bhh  = (const float*)d_in[5];   // [G4H]
    const float* Wout = (const float*)d_in[6];   // [V_, H_]
    const float* bout = (const float*)d_in[7];   // [V_]

    // Workspace carve-out (256B aligned slices)
    char*  ws  = (char*)d_ws;
    size_t off = 0;
    auto carve = [&](size_t bytes) -> void* {
        void* p = ws + off;
        off += (bytes + 255) & ~(size_t)255;
        return p;
    };
    __bf16* WihB  = (__bf16*)carve((size_t)G4H * E_ * 2);
    __bf16* WhhB  = (__bf16*)carve((size_t)G4H * H_ * 2);
    __bf16* WoutB = (__bf16*)carve((size_t)V_  * H_ * 2);
    __bf16* embB  = (__bf16*)carve((size_t)V_  * E_ * 2);
    float*  cst   = (float*) carve((size_t)B_ * H_ * 4);
    __bf16* hB    = (__bf16*)carve((size_t)B_ * H_ * 2);
    __bf16* xB    = (__bf16*)carve((size_t)B_ * E_ * 2);
    float*  gates = (float*) carve((size_t)B_ * G4H * 4);
    (void)ws_size; (void)in_sizes; (void)n_in; (void)out_size;

    // Output layout: tokens [T,B] int32 first, then logits [T,B,V] f32
    int*   tokens     = (int*)d_out;
    float* logits_out = (float*)d_out + (size_t)TSTEPS * B_;

    // --- one-time conversions / state init ---
    {
        int n;
        n = G4H * E_;       f32_to_bf16_kernel<<<n / 256, 256, 0, stream>>>(Wih,  WihB,  n);
        n = G4H * H_;       f32_to_bf16_kernel<<<n / 256, 256, 0, stream>>>(Whh,  WhhB,  n);
        n = V_ * H_;        f32_to_bf16_kernel<<<n / 256, 256, 0, stream>>>(Wout, WoutB, n);
        n = V_ * E_;        f32_to_bf16_kernel<<<n / 256, 256, 0, stream>>>(emb,  embB,  n);
        init_state_kernel<<<(B_ * H_) / 256, 256, 0, stream>>>(obs, cst, hB);
        init_x_kernel<<<(B_ * E_) / 256, 256, 0, stream>>>(embB, xB);
    }

    // --- 200 sequential decode steps ---
    for (unsigned t = 0; t < TSTEPS; ++t) {
        gates_kernel<<<G4H / 64, 256, 0, stream>>>(xB, hB, WihB, WhhB, bih, bhh, gates);
        lstm_update_kernel<<<(B_ * H_) / 256, 256, 0, stream>>>(gates, cst, hB);
        float* lg = logits_out + (size_t)t * B_ * V_;
        logits_kernel<<<V_ / 64, 256, 0, stream>>>(hB, WoutB, bout, lg);
        sample_gather_kernel<<<B_, 256, 0, stream>>>(lg, embB, tokens + (size_t)t * B_, xB, t);
    }
}
